// CodeEntropyLoss_72499047957160
// MI455X (gfx1250) — matrix operand order repaired and checked by reading, not compiled, not run on hardware
//
#include <hip/hip_runtime.h>
#include <hip/hip_bf16.h>

// ---------------------------------------------------------------------------
// CodeEntropyLoss fused kernel for MI455X (gfx1250, wave32, WMMA).
//
// distance-GEMM: N=B*T (24000) x K=4096 x D=512, bf16 WMMA / f32 accum,
// fused softmax via 256KB LDS logits staging (CDNA5: 320KB LDS/WGP),
// register-resident A tile + software-pipelined coalesced B stream.
// ---------------------------------------------------------------------------

constexpr int KC    = 4096;        // NUM_CODES
constexpr int DD    = 512;         // feature dim
constexpr int STEPS = DD / 32;     // 16 WMMA K-steps per tile
constexpr int PK    = KC + 4;      // logits row pad: 8*4100 % 64 == 32 -> no
                                   // bank aliasing between half-wave groups

typedef __attribute__((ext_vector_type(16))) __bf16 v16bf;
typedef __attribute__((ext_vector_type(8)))  float  v8f;

union Frag {
    uint4 u[2];   // 32 bytes = 16 bf16
    v16bf v;
};

__device__ __forceinline__ unsigned short f32_to_bf16(float f) {
    unsigned u = __float_as_uint(f);
    u += 0x7fffu + ((u >> 16) & 1u);   // round-to-nearest-even
    return (unsigned short)(u >> 16);
}

// ---------------------------------------------------------------------------
// Kernel 1: codebook fp32 -> packed bf16 B-fragments + ||c||^2.
// Packed layout: [tile][step][lane][16 bf16] so a wave's B fetch for one
// WMMA step is a single contiguous, fully-coalesced 1KB burst.
//   lane = half*16 + q ; chunk = codebook[tile*16+q][step*32 + half*16 .. +15]
// ---------------------------------------------------------------------------
__global__ void prep_codebook(const float* __restrict__ cb,
                              unsigned short* __restrict__ pkB,
                              float* __restrict__ c2) {
    const int k    = blockIdx.x;          // code id
    const int lane = threadIdx.x;         // 32 threads
    const float* src = cb + (size_t)k * DD;
    const int tt = k >> 4, q = k & 15;
    float s = 0.f;
    for (int d = lane; d < DD; d += 32) {
        float f = src[d];
        s += f * f;
        const int st = d >> 5, rem = d & 31, hf = rem >> 4, j = rem & 15;
        const int ln = hf * 16 + q;
        pkB[(((size_t)(tt * 16 + st) * 32) + ln) * 16 + j] = f32_to_bf16(f);
    }
    for (int off = 16; off; off >>= 1) s += __shfl_xor(s, off, 32);
    if (lane == 0) c2[k] = s;
}

// ---------------------------------------------------------------------------
// Kernel 2: fused distance-GEMM + softmax + masked histogram.
// grid = N/16 blocks, 256 threads (8 waves). Dynamic LDS:
//   [ logits f32 16 x PK | A bf16 16 x DD | z2s[16] rowmax[16] rowscale[16] maskv[16] ]
// ---------------------------------------------------------------------------
__global__ __launch_bounds__(256)
void entropy_main(const float* __restrict__ sf,             // (B, D, T) fp32
                  const unsigned short* __restrict__ pkB,   // packed bf16 B
                  const float* __restrict__ c2g,            // (K)
                  const int* __restrict__ lengths,          // (B)
                  const int* __restrict__ stridep,          // scalar
                  float* __restrict__ cdist,                // (K) accumulator
                  int Tv, int Ntot) {
    extern __shared__ char smem[];
    float*          L        = (float*)smem;                                  // 16*PK
    unsigned short* Abuf     = (unsigned short*)(smem + (size_t)16 * PK * 4); // 16*DD
    float*          z2s      = (float*)(Abuf + (size_t)16 * DD);              // 16
    float*          rowmax   = z2s + 16;
    float*          rowscale = rowmax + 16;
    float*          maskv    = rowscale + 16;

    const int tid     = threadIdx.x;
    const int rowbase = blockIdx.x * 16;

    // ---- length mask for the 16 rows; early-out if fully masked ----------
    int validFlag = 0;
    if (tid < 16) {
        z2s[tid] = 0.f;
        int row = rowbase + tid;
        float mk = 0.f;
        if (row < Ntot) {
            int b = row / Tv;
            int t = row - b * Tv;
            int nv = lengths[b] / stridep[0];
            if (nv > Tv) nv = Tv;
            if (t < nv) mk = 1.f;
        }
        maskv[tid] = mk;
        validFlag = (mk > 0.f);
    }
    if (!__syncthreads_or(validFlag)) return;   // whole tile masked: skip GEMM

    // ---- load A tile (16 rows of z), convert to bf16, accumulate ||z||^2 --
    {
        const int m   = tid & 15;
        const int row = rowbase + m;
        float part = 0.f;
        if (row < Ntot) {
            int b = row / Tv;
            int t = row - b * Tv;
            const float* src = sf + (size_t)b * DD * Tv + t;   // stride Tv along d
            for (int d = tid >> 4; d < DD; d += 16) {
                float f = src[(size_t)d * Tv];
                part += f * f;
                Abuf[m * DD + d] = f32_to_bf16(f);
            }
        } else {
            for (int d = tid >> 4; d < DD; d += 16) Abuf[m * DD + d] = 0;
        }
        atomicAdd(&z2s[m], part);
    }
    __syncthreads();

    const int wave   = tid >> 5;
    const int lane   = tid & 31;
    const int half   = lane >> 4;      // which half-wave (K split / M split)
    const int q      = lane & 15;      // A row index / B column index
    const int nwaves = blockDim.x >> 5;
    const int ntiles = KC >> 4;        // 256 column tiles

    // ---- A fragments: register-resident for the whole tile loop ----------
    // 16-bit A 16x32 layout: elems 0..15 -> K = kb+16*half.. (contiguous, matches packed B)
    Frag afr[STEPS];
    {
        const unsigned short* aptr = Abuf + q * DD + 16 * half;
#pragma unroll
        for (int s = 0; s < STEPS; ++s) {
            afr[s].u[0] = *(const uint4*)(aptr + 32 * s);        // ds_load_b128
            afr[s].u[1] = *(const uint4*)(aptr + 32 * s + 8);
        }
    }
    // per-row ||z||^2 for this lane's C rows (r + 8*half)
    float z2r[8];
#pragma unroll
    for (int r = 0; r < 8; ++r) z2r[r] = z2s[r + 8 * half];

    // ---- distance GEMM: pipelined B stream, 4-deep rotating fragments ----
    constexpr size_t TILE_SPAN = (size_t)STEPS * 32 * 16;   // ushorts per tile
    int tile = wave;
    const unsigned short* bl = pkB + (size_t)tile * TILE_SPAN + (size_t)lane * 16;
    Frag bf4[4];
#pragma unroll
    for (int s = 0; s < 4; ++s) {                            // prime pipeline
        bf4[s].u[0] = *(const uint4*)(bl + s * 512);
        bf4[s].u[1] = *(const uint4*)(bl + s * 512 + 8);
    }
    for (; tile < ntiles; tile += nwaves) {
        const int nt = tile + nwaves;
        const unsigned short* nbl =
            pkB + (size_t)((nt < ntiles) ? nt : wave) * TILE_SPAN + (size_t)lane * 16;
        v8f acc = {0.f, 0.f, 0.f, 0.f, 0.f, 0.f, 0.f, 0.f};
#pragma unroll
        for (int s = 0; s < STEPS; ++s) {
            acc = __builtin_amdgcn_wmma_f32_16x16x32_bf16(
                false, afr[s].v, false, bf4[s & 3].v, (short)0, acc, false, false);
            // refill the slot just consumed; roll into next tile at the tail
            const unsigned short* p =
                (s < STEPS - 4) ? (bl + (s + 4) * 512) : (nbl + (s - (STEPS - 4)) * 512);
            bf4[s & 3].u[0] = *(const uint4*)p;
            bf4[s & 3].u[1] = *(const uint4*)(p + 8);
        }
        // C/D layout: acc[r] -> row (r + 8*half), col q
        const int cb0 = tile << 4;
        const float c2n = c2g[cb0 + q];
#pragma unroll
        for (int r = 0; r < 8; ++r) {
            float d2 = z2r[r] + c2n - 2.f * acc[r];
            L[(size_t)(r + 8 * half) * PK + cb0 + q] = -sqrtf(fmaxf(d2, 1e-12f));
        }
        bl = nbl;
    }
    __syncthreads();

    // ---- per-row softmax stats (one wave per row, wave32 shuffles) --------
    for (int m = wave; m < 16; m += nwaves) {
        const float* Lr = L + (size_t)m * PK;
        float mx = -3.4e38f;
        for (int k = lane; k < KC; k += 32) mx = fmaxf(mx, Lr[k]);
        for (int off = 16; off; off >>= 1) mx = fmaxf(mx, __shfl_xor(mx, off, 32));
        float s = 0.f;
        for (int k = lane; k < KC; k += 32) s += __expf(Lr[k] - mx);
        for (int off = 16; off; off >>= 1) s += __shfl_xor(s, off, 32);
        if (lane == 0) {
            rowmax[m]   = mx;
            rowscale[m] = maskv[m] / s;   // fold mask into softmax scale
        }
    }
    __syncthreads();

    // ---- masked probability histogram: one atomic per code per block ------
    for (int k = tid; k < KC; k += blockDim.x) {
        float s = 0.f;
        for (int m = 0; m < 16; ++m) {
            float sc = rowscale[m];
            if (sc != 0.f) s += __expf(L[(size_t)m * PK + k] - rowmax[m]) * sc;
        }
        if (s != 0.f) atomicAdd(&cdist[k], s);
    }
}

// ---------------------------------------------------------------------------
// Kernel 3: normalize histogram, entropy, scalar output.
// ---------------------------------------------------------------------------
__global__ void entropy_finalize(const float* __restrict__ cd,
                                 float* __restrict__ out) {
    __shared__ float red[256];
    __shared__ float totS;
    const int tid = threadIdx.x;
    float t = 0.f;
    for (int k = tid; k < KC; k += 256) t += cd[k];
    red[tid] = t;
    __syncthreads();
    for (int s = 128; s; s >>= 1) {
        if (tid < s) red[tid] += red[tid + s];
        __syncthreads();
    }
    if (tid == 0) totS = red[0] + 1e-8f;
    __syncthreads();
    const float tot = totS;
    float e = 0.f;
    for (int k = tid; k < KC; k += 256) {
        float p = cd[k] / tot;
        e -= p * logf(p + 1e-8f);
    }
    red[tid] = e;
    __syncthreads();
    for (int s = 128; s; s >>= 1) {
        if (tid < s) red[tid] += red[tid + s];
        __syncthreads();
    }
    if (tid == 0) out[0] = 1.f - red[0] / logf((float)KC);
}

// ---------------------------------------------------------------------------
extern "C" void kernel_launch(void* const* d_in, const int* in_sizes, int n_in,
                              void* d_out, int out_size, void* d_ws, size_t ws_size,
                              hipStream_t stream) {
    const float* sf      = (const float*)d_in[0];   // (B, D, T)
    const float* cb      = (const float*)d_in[1];   // (K, D)
    const int*   lengths = (const int*)d_in[2];     // (B)
    const int*   stridep = (const int*)d_in[3];     // scalar

    const int Bv   = in_sizes[2];
    const int Tv   = in_sizes[0] / (Bv * DD);       // 1500
    const int Ntot = Bv * Tv;                       // 24000

    // workspace: [ packed bf16 B (K*D) | c2 (K f32) | code_dist (K f32) ]
    unsigned short* pkB = (unsigned short*)d_ws;
    float* c2    = (float*)((char*)d_ws + (size_t)KC * DD * sizeof(unsigned short));
    float* cdist = c2 + KC;

    hipMemsetAsync(cdist, 0, KC * sizeof(float), stream);

    prep_codebook<<<KC, 32, 0, stream>>>(cb, pkB, c2);

    const size_t smem = (size_t)16 * PK * sizeof(float)          // logits
                      + (size_t)16 * DD * sizeof(unsigned short) // A tile
                      + 64 * sizeof(float);                      // stats
    (void)hipFuncSetAttribute((const void*)entropy_main,
                              hipFuncAttributeMaxDynamicSharedMemorySize,
                              (int)smem);
    const int nblocks = (Ntot + 15) / 16;
    entropy_main<<<nblocks, 256, smem, stream>>>(
        sf, pkB, c2, lengths, stridep, cdist, Tv, Ntot);

    entropy_finalize<<<1, 256, 0, stream>>>(cdist, (float*)d_out);
}
